// GRUUpdate_88579405512822
// MI455X (gfx1250) — matrix operand order repaired and checked by reading, not compiled, not run on hardware
//
#include <hip/hip_runtime.h>
#include <hip/hip_bf16.h>

typedef __attribute__((ext_vector_type(16))) __bf16 v16bf;
typedef __attribute__((ext_vector_type(8)))  float  v8f;

#define B_   32
#define N_   4096
#define E_   8192
#define D_   128
#define COLS 384   // 3*D

static __device__ __forceinline__ unsigned short f2bf(float x) {
    unsigned u = __float_as_uint(x);
    u += 0x7FFFu + ((u >> 16) & 1u);          // round-to-nearest-even
    return (unsigned short)(u >> 16);
}

__global__ __launch_bounds__(256) void zero_agg(float4* __restrict__ p, int n4) {
    int t = blockIdx.x * blockDim.x + threadIdx.x;
    if (t < n4) p[t] = make_float4(0.f, 0.f, 0.f, 0.f);
}

// Repack fp32 weights (128 x 384 row-major) into bf16 fragment-major layout:
// element index = ((colTile*4 + kTile)*32 + lane)*16 + j
// value = W[(kTile*32 + (lane>>4)*16 + j)*384 + colTile*16 + (lane&15)]
__global__ __launch_bounds__(256) void repack_weights(const float* __restrict__ Wk,
                                                      const float* __restrict__ Wr,
                                                      unsigned short* __restrict__ Ok,
                                                      unsigned short* __restrict__ Or) {
    int t = blockIdx.x * blockDim.x + threadIdx.x;     // 2 * 49152 threads
    int mat = t / (D_ * COLS);
    int i   = t % (D_ * COLS);
    int j       = i & 15;
    int lane    = (i >> 4) & 31;
    int kt      = (i >> 9) & 3;
    int colTile = i >> 11;                              // 0..23
    int hi  = lane >> 4;
    int k   = kt * 32 + hi * 16 + j;
    int col = colTile * 16 + (lane & 15);
    const float* src = mat ? Wr : Wk;
    unsigned short* dst = mat ? Or : Ok;
    dst[i] = f2bf(src[(size_t)k * COLS + col]);
}

// Scatter-add: aggregated[b, tgt, :] += messages[b, e, :]
__global__ __launch_bounds__(256) void scatter_msgs(const float* __restrict__ msgs,
                                                    const int* __restrict__ conn,
                                                    float* __restrict__ agg) {
    int t  = blockIdx.x * blockDim.x + threadIdx.x;     // B*E*32 threads
    int eg = t >> 5;                                    // flat edge index b*E+e
    int d4 = (t & 31) << 2;
    int b  = eg >> 13;                                  // E = 8192 = 2^13
    int tgt = conn[(eg << 1) + 1];
    const float* m = msgs + (size_t)eg * D_ + d4;
    float* a = agg + ((size_t)(b * N_ + tgt) * D_ + d4);
    unsafeAtomicAdd(a + 0, m[0]);
    unsafeAtomicAdd(a + 1, m[1]);
    unsafeAtomicAdd(a + 2, m[2]);
    unsafeAtomicAdd(a + 3, m[3]);
}

// Fused dual-GEMM + GRU gate epilogue. One block = 16 rows, 8 waves.
// Wave w owns output columns [16w, 16w+16): column tiles w (z), w+8 (r), w+16 (h).
__global__ __launch_bounds__(256) void gru_fused(const float* __restrict__ agg,
                                                 const float* __restrict__ hmat,
                                                 const unsigned short* __restrict__ WkBf,
                                                 const unsigned short* __restrict__ WrBf,
                                                 const float* __restrict__ bias,
                                                 float* __restrict__ out) {
    __shared__ unsigned short ldsA[2 * 2048];   // [mat][(kt*32+lane)*16 + j], bf16 bits
    const int tid = threadIdx.x;
    const int rowBase = blockIdx.x * 16;

    // Stage aggregated (mat=0) and atom_state (mat=1) 16x128 tiles into LDS,
    // converted to bf16, already in the A-fragment k order this code uses (contiguous k, same as B):
    //   lane<16: row M=lane, K = hi*16 + j; lane>=16: same with hi=1.
    for (int i = tid; i < 4096; i += 256) {
        int j    = i & 15;
        int lane = (i >> 4) & 31;
        int kt   = (i >> 9) & 3;
        int mat  = i >> 11;
        int hi   = lane >> 4;
        int m    = lane & 15;
        int k    = kt * 32 + hi * 16 + j;
        const float* src = mat ? hmat : agg;
        ldsA[i] = f2bf(src[(size_t)(rowBase + m) * D_ + k]);
    }
    __syncthreads();

    const int wave = tid >> 5;
    const int lane = tid & 31;
    const int hi   = lane >> 4;
    const int n    = lane & 15;

    const v16bf* Ax = (const v16bf*)ldsA;           // fragment index: kt*32 + lane
    const v16bf* Ah = (const v16bf*)(ldsA + 2048);
    const v16bf* Bk = (const v16bf*)WkBf;           // fragment index: (ct*4+kt)*32 + lane
    const v16bf* Br = (const v16bf*)WrBf;

    v8f c_xz = {}, c_hz = {}, c_xr = {}, c_hr = {}, c_xh = {}, c_hh = {};

#pragma unroll
    for (int kt = 0; kt < 4; ++kt) {
        v16bf ax = Ax[kt * 32 + lane];
        v16bf ah = Ah[kt * 32 + lane];

        int fz = ((0 * 8 + wave) * 4 + kt) * 32 + lane;
        int fr = ((1 * 8 + wave) * 4 + kt) * 32 + lane;
        int fh = ((2 * 8 + wave) * 4 + kt) * 32 + lane;

        c_xz = __builtin_amdgcn_wmma_f32_16x16x32_bf16(false, ax, false, Bk[fz], (short)0, c_xz, false, false);
        c_hz = __builtin_amdgcn_wmma_f32_16x16x32_bf16(false, ah, false, Br[fz], (short)0, c_hz, false, false);
        c_xr = __builtin_amdgcn_wmma_f32_16x16x32_bf16(false, ax, false, Bk[fr], (short)0, c_xr, false, false);
        c_hr = __builtin_amdgcn_wmma_f32_16x16x32_bf16(false, ah, false, Br[fr], (short)0, c_hr, false, false);
        c_xh = __builtin_amdgcn_wmma_f32_16x16x32_bf16(false, ax, false, Bk[fh], (short)0, c_xh, false, false);
        c_hh = __builtin_amdgcn_wmma_f32_16x16x32_bf16(false, ah, false, Br[fh], (short)0, c_hh, false, false);
    }

    // GRU epilogue. C/D layout: VGPR v holds (M = v + 8*hi, N = lane&15).
    const int cz = wave * 16 + n;                   // output column in [0,128)
    const float b0z = bias[cz],        b1z = bias[COLS + cz];
    const float b0r = bias[128 + cz],  b1r = bias[COLS + 128 + cz];
    const float b0h = bias[256 + cz],  b1h = bias[COLS + 256 + cz];

#pragma unroll
    for (int v = 0; v < 8; ++v) {
        int m = v + hi * 8;
        size_t ri = (size_t)(rowBase + m) * D_ + cz;
        float hval = hmat[ri];
        float zz = c_xz[v] + b0z + c_hz[v] + b1z;
        float rr = c_xr[v] + b0r + c_hr[v] + b1r;
        float z  = 1.f / (1.f + __expf(-zz));
        float r  = 1.f / (1.f + __expf(-rr));
        float pre = c_xh[v] + b0h + r * (c_hh[v] + b1h);
        float th  = 1.f - 2.f / (__expf(2.f * pre) + 1.f);   // tanh(pre)
        out[ri] = z * hval + (1.f - z) * th;
    }
}

extern "C" void kernel_launch(void* const* d_in, const int* in_sizes, int n_in,
                              void* d_out, int out_size, void* d_ws, size_t ws_size,
                              hipStream_t stream) {
    const float* atom_state = (const float*)d_in[0];
    const float* messages   = (const float*)d_in[1];
    const int*   conn       = (const int*)d_in[2];
    const float* Wk         = (const float*)d_in[3];
    const float* Wr         = (const float*)d_in[4];
    const float* bias       = (const float*)d_in[5];
    float* out = (float*)d_out;

    float* agg = (float*)d_ws;
    size_t aggBytes = (size_t)B_ * N_ * D_ * sizeof(float);          // 64 MB
    unsigned short* WkBf = (unsigned short*)((char*)d_ws + aggBytes);
    unsigned short* WrBf = WkBf + (size_t)D_ * COLS;

    int n4 = (B_ * N_ * D_) / 4;                                     // 4,194,304
    zero_agg<<<n4 / 256, 256, 0, stream>>>((float4*)agg, n4);
    repack_weights<<<(2 * D_ * COLS) / 256, 256, 0, stream>>>(Wk, Wr, WkBf, WrBf);
    scatter_msgs<<<(B_ * E_ * 32) / 256, 256, 0, stream>>>(messages, conn, agg);
    gru_fused<<<(B_ * N_) / 16, 256, 0, stream>>>(agg, atom_state, WkBf, WrBf, bias, out);
}
